// AKTNet_37469294690345
// MI455X (gfx1250) — matrix operand order repaired and checked by reading, not compiled, not run on hardware
//
#include <hip/hip_runtime.h>
#include <hip/hip_bf16.h>
#include <cmath>

#define BS_     8
#define SEQ     512
#define DMODEL  256
#define NHEADS  8
#define DK_     32
#define DFF     2048
#define FC1_    512
#define NQ      10000
#define ROWS    (BS_*SEQ)   /* 4096 */

typedef __attribute__((ext_vector_type(16))) __bf16 v16bf;
typedef __attribute__((ext_vector_type(8)))  float  v8f;

static __device__ __forceinline__ float warp_sum(float v) {
#pragma unroll
  for (int off = 16; off > 0; off >>= 1) v += __shfl_xor(v, off, 32);
  return v;
}
static __device__ __forceinline__ float warp_max(float v) {
#pragma unroll
  for (int off = 16; off > 0; off >>= 1) v = fmaxf(v, __shfl_xor(v, off, 32));
  return v;
}

// gfx1250 async global->LDS copy (ASYNCcnt-tracked, VGPR-bypassing).
// lds_off = byte offset into the wave's LDS (low 32 bits of a generic
// pointer into __shared__), gaddr = 64-bit global address.
static __device__ __forceinline__ void async_copy_b128(uint32_t lds_off,
                                                       const float* gaddr) {
  asm volatile("global_load_async_to_lds_b128 %0, %1, off"
               :: "v"(lds_off), "v"(gaddr) : "memory");
}
static __device__ __forceinline__ void wait_async_le2() {
  asm volatile("s_wait_asynccnt 0x2" ::: "memory");
}
static __device__ __forceinline__ void wait_async_0() {
  asm volatile("s_wait_asynccnt 0x0" ::: "memory");
}

// ---------------------------------------------------------------------------
// GEMM: C[M,N] = act(A[M,K] @ B[K,N] + bias), fp32 in/out, bf16 WMMA accum f32.
// Block = 256 threads (8 waves), block tile = 128 rows x 64 cols:
//   wave w -> M-tile (blockIdx.y*8 + w), 4 N-tiles (A fragment reused 4x).
// B panel (32 x 64 f32, 8KB) staged in LDS by async b128 copies, triple
// buffered, pipelined with s_wait_asynccnt (async loads complete in order).
// Requires M%128==0, N%64==0, K%32==0 (true for all call sites here).
// VGPR layouts used here (A/B in contiguous k order, not the ISA table order):
//   A 16-bit 16x32: lane=(M=lane&15, half=lane>>4); elem j -> K = half*16 + j
//   B 16-bit 32x16: lane=(N=lane&15, half=lane>>4); elem j -> K = half*16 + j
//   C/D f32 16x16:  elem r -> row r + half*8, col lane&15
// ---------------------------------------------------------------------------
__global__ __launch_bounds__(256) void gemm_bf16_wmma(
    const float* __restrict__ A, const float* __restrict__ B,
    const float* __restrict__ bias, float* __restrict__ C,
    int M, int N, int K, int relu) {
  __shared__ float bpan[3][32 * 64];            // 24 KB, triple buffered
  const int wave = threadIdx.x >> 5;
  const int lane = threadIdx.x & 31;
  const int row  = lane & 15;
  const int half = lane >> 4;
  const int n0   = blockIdx.x * 64;
  const int mt   = blockIdx.y * 8 + wave;

  // staging role of this thread: 8 consecutive f32 of the 32x64 panel
  const int se = threadIdx.x * 8;               // 0..2047
  const int sk = se >> 6;                       // panel row (k)
  const int sn = se & 63;                       // panel col (n)
  const uint32_t lds_base =
      (uint32_t)(uintptr_t)(&bpan[0][0]) + (uint32_t)(se * 4);
  const float* gbase = B + (size_t)sk * N + n0 + sn;

  const int nsteps = K >> 5;
  async_copy_b128(lds_base, gbase);             // prefetch panel 0
  async_copy_b128(lds_base + 16, gbase + 4);

  const float* Arow = A + (size_t)(mt * 16 + row) * K;
  v8f acc0 = {}, acc1 = {}, acc2 = {}, acc3 = {};

  for (int step = 0; step < nsteps; ++step) {
    const int buf = step % 3;
    if (step + 1 < nsteps) {                    // prefetch next panel
      const int nbuf = (step + 1) % 3;
      const float* gp = gbase + (size_t)(step + 1) * 32 * N;
      async_copy_b128(lds_base + nbuf * 8192, gp);
      async_copy_b128(lds_base + nbuf * 8192 + 16, gp + 4);
      wait_async_le2();                         // current panel resident
    } else {
      wait_async_0();
    }
    __syncthreads();

    // A fragment (direct from global; rows are block-exclusive)
    const int kb = step * 32 + half * 16;
    v16bf a;
#pragma unroll
    for (int j = 0; j < 8; ++j)  a[j] = (__bf16)Arow[kb + j];
#pragma unroll
    for (int j = 8; j < 16; ++j) a[j] = (__bf16)Arow[kb + j];

    // 4 B fragments from the LDS panel (convert f32->bf16 on read)
    const float* P = &bpan[buf][0];
    v16bf b0, b1, b2, b3;
#pragma unroll
    for (int j = 0; j < 16; ++j) {
      const float* pr = P + (half * 16 + j) * 64 + row;
      b0[j] = (__bf16)pr[0];
      b1[j] = (__bf16)pr[16];
      b2[j] = (__bf16)pr[32];
      b3[j] = (__bf16)pr[48];
    }
    acc0 = __builtin_amdgcn_wmma_f32_16x16x32_bf16(false, a, false, b0,
                                                   (short)0, acc0, false, false);
    acc1 = __builtin_amdgcn_wmma_f32_16x16x32_bf16(false, a, false, b1,
                                                   (short)0, acc1, false, false);
    acc2 = __builtin_amdgcn_wmma_f32_16x16x32_bf16(false, a, false, b2,
                                                   (short)0, acc2, false, false);
    acc3 = __builtin_amdgcn_wmma_f32_16x16x32_bf16(false, a, false, b3,
                                                   (short)0, acc3, false, false);
  }

#pragma unroll
  for (int r = 0; r < 8; ++r) {
    const int m = mt * 16 + r + half * 8;
    float* crow = C + (size_t)m * N + n0 + row;
    float v0 = acc0[r] + bias[n0 + row];
    float v1 = acc1[r] + bias[n0 + 16 + row];
    float v2 = acc2[r] + bias[n0 + 32 + row];
    float v3 = acc3[r] + bias[n0 + 48 + row];
    if (relu) {
      v0 = fmaxf(v0, 0.f); v1 = fmaxf(v1, 0.f);
      v2 = fmaxf(v2, 0.f); v3 = fmaxf(v3, 0.f);
    }
    crow[0] = v0; crow[16] = v1; crow[32] = v2; crow[48] = v3;
  }
}

// ---------------------------------------------------------------------------
// Embedding / gather kernel: builds q_e, x = q_e+pos, y = qa_e+pos, g.
// ---------------------------------------------------------------------------
__global__ __launch_bounds__(256) void embed_kernel(
    const int* __restrict__ q_data, const int* __restrict__ qa_data,
    const int* __restrict__ graph,
    const float* __restrict__ q_embed, const float* __restrict__ qa_embed,
    const float* __restrict__ pos_embed, const float* __restrict__ graph_embed,
    float* __restrict__ q_e, float* __restrict__ x,
    float* __restrict__ y, float* __restrict__ g) {
  const int idx = blockIdx.x;         // b*SEQ + s
  const int b = idx / SEQ, s = idx % SEQ;
  const int d = threadIdx.x;
  const int q = q_data[idx];
  const int r = (qa_data[idx] - q) / NQ;          // response in {0,1}
  const int gi = graph[s * BS_ + b];              // graph is (S, BS)
  const float qe = q_embed[(size_t)q * DMODEL + d];
  const float pe = pos_embed[(size_t)s * DMODEL + d];
  const size_t o = (size_t)idx * DMODEL + d;
  q_e[o] = qe;
  x[o]   = qe + pe;
  y[o]   = qa_embed[(size_t)r * DMODEL + d] + qe + pe;
  g[o]   = graph_embed[(size_t)gi * DMODEL + d];
}

// ---------------------------------------------------------------------------
// AKT attention: one block per (query 16-row tile, head, batch).
// blockDim = 256 (8 waves). LDS: scores/attn 16x512 f32 + 8 partial 16x32.
// ---------------------------------------------------------------------------
__global__ __launch_bounds__(256) void attn_kernel(
    const float* __restrict__ Qp, const float* __restrict__ Kp,
    const float* __restrict__ Vp, const float* __restrict__ gammas,
    float* __restrict__ Out, int mask_flag) {
  __shared__ float sc[16][SEQ];          // 32 KB: scores, then attn weights
  __shared__ float red[8][16][DK_];      // 16 KB: per-wave attn@V partials

  const int mblk = blockIdx.x;           // 0..31
  const int h    = blockIdx.y;
  const int b    = blockIdx.z;
  const int wave = threadIdx.x >> 5;
  const int lane = threadIdx.x & 31;
  const int row  = lane & 15;
  const int half = lane >> 4;
  const float scale = 0.17677669529663687f;   // 1/sqrt(DK)

  const float* Qb = Qp + (size_t)b * SEQ * DMODEL + h * DK_;
  const float* Kb = Kp + (size_t)b * SEQ * DMODEL + h * DK_;
  const float* Vb = Vp + (size_t)b * SEQ * DMODEL + h * DK_;

  // ---- 1) scores = (Q @ K^T) * scale : one WMMA per 16x16 tile (K-dim = 32)
  v16bf aQ;
  {
    const float* qrow = Qb + (size_t)(mblk * 16 + row) * DMODEL;
#pragma unroll
    for (int j = 0; j < 8; ++j)  aQ[j] = (__bf16)qrow[half * 16 + j];
#pragma unroll
    for (int j = 8; j < 16; ++j) aQ[j] = (__bf16)qrow[half * 16 + j];
  }
  for (int tt = wave; tt < SEQ / 16; tt += 8) {
    v16bf bK;
    const float* krow = Kb + (size_t)(tt * 16 + row) * DMODEL;
#pragma unroll
    for (int j = 0; j < 16; ++j) bK[j] = (__bf16)krow[half * 16 + j];
    v8f s = {};
    s = __builtin_amdgcn_wmma_f32_16x16x32_bf16(false, aQ, false, bK,
                                                (short)0, s, false, false);
#pragma unroll
    for (int r = 0; r < 8; ++r)
      sc[r + half * 8][tt * 16 + row] = s[r] * scale;
  }
  __syncthreads();

  // ---- 2) per-row: masked softmax -> cumsum -> distance effect -> softmax
  const float gg = gammas[h];
  const float sp = (gg > 20.f) ? gg : log1pf(__expf(gg));   // softplus
  const float gamma = -sp;
  const int zero_pad = (mask_flag == 0);

  for (int rr = 0; rr < 2; ++rr) {
    const int r = wave * 2 + rr;
    const int i = mblk * 16 + r;          // global query index
    const int jmax = i + mask_flag - 1;   // attend iff j <= jmax
    float v[16];
#pragma unroll
    for (int c = 0; c < 16; ++c) v[c] = sc[r][c * 32 + lane];

    float m1 = -3.0e38f;
#pragma unroll
    for (int c = 0; c < 16; ++c) {
      const int j = c * 32 + lane;
      if (j <= jmax) m1 = fmaxf(m1, v[c]);
    }
    m1 = warp_max(m1);

    float e[16]; float s1 = 0.f;
#pragma unroll
    for (int c = 0; c < 16; ++c) {
      const int j = c * 32 + lane;
      e[c] = (j <= jmax) ? __expf(v[c] - m1) : 0.f;
      s1 += e[c];
    }
    s1 = warp_sum(s1);
    const float inv1 = (s1 > 0.f) ? 1.f / s1 : 0.f;

    // inclusive cumsum of the masked softmax along the row (chunks of 32, wave scan + carry)
    float cum[16]; float carry = 0.f;
    for (int c = 0; c < 16; ++c) {
      float yv = e[c] * inv1;
#pragma unroll
      for (int off = 1; off < 32; off <<= 1) {
        const float t = __shfl_up(yv, off, 32);
        if (lane >= off) yv += t;
      }
      cum[c] = yv + carry;
      carry += __shfl(yv, 31, 32);
    }
    const float disttot = carry;

    float att[16]; float m2 = -3.0e38f;
#pragma unroll
    for (int c = 0; c < 16; ++c) {
      const int j = c * 32 + lane;
      const float pos  = fabsf((float)(i - j));
      const float dist = sqrtf(fmaxf((disttot - cum[c]) * pos, 0.f));
      float eff = __expf(dist * gamma);
      eff = fminf(fmaxf(eff, 1e-5f), 1e5f);
      att[c] = v[c] * eff;                 // second softmax is UNMASKED
      m2 = fmaxf(m2, att[c]);
    }
    m2 = warp_max(m2);
    float s2 = 0.f;
#pragma unroll
    for (int c = 0; c < 16; ++c) { att[c] = __expf(att[c] - m2); s2 += att[c]; }
    s2 = warp_sum(s2);
    const float inv2 = 1.f / s2;
#pragma unroll
    for (int c = 0; c < 16; ++c) {
      float a = att[c] * inv2;
      if (zero_pad && i == 0) a = 0.f;     // attn[:, :, 0, :] = 0
      sc[r][c * 32 + lane] = a;
    }
  }
  __syncthreads();

  // ---- 3) out = attn @ V : K=512 split over 8 waves (64 each), 2 N-tiles
  v8f o0 = {}, o1 = {};
  for (int kc = 0; kc < 2; ++kc) {
    const int kk = wave * 64 + kc * 32;
    v16bf a, b0, b1;
#pragma unroll
    for (int j = 0; j < 8; ++j)  a[j] = (__bf16)sc[row][kk + half * 16 + j];
#pragma unroll
    for (int j = 8; j < 16; ++j) a[j] = (__bf16)sc[row][kk + half * 16 + j];
#pragma unroll
    for (int j = 0; j < 16; ++j) {
      const float* vrow = Vb + (size_t)(kk + half * 16 + j) * DMODEL;
      b0[j] = (__bf16)vrow[row];
      b1[j] = (__bf16)vrow[16 + row];
    }
    o0 = __builtin_amdgcn_wmma_f32_16x16x32_bf16(false, a, false, b0,
                                                 (short)0, o0, false, false);
    o1 = __builtin_amdgcn_wmma_f32_16x16x32_bf16(false, a, false, b1,
                                                 (short)0, o1, false, false);
  }
#pragma unroll
  for (int r = 0; r < 8; ++r) {
    red[wave][r + half * 8][row]      = o0[r];
    red[wave][r + half * 8][16 + row] = o1[r];
  }
  __syncthreads();

  for (int idx = threadIdx.x; idx < 16 * DK_; idx += 256) {
    const int r = idx >> 5, d = idx & 31;
    float acc = 0.f;
#pragma unroll
    for (int w = 0; w < 8; ++w) acc += red[w][r][d];
    Out[((size_t)b * SEQ + mblk * 16 + r) * DMODEL + h * DK_ + d] = acc;
  }
}

// ---------------------------------------------------------------------------
// LayerNorm of (a + b): one wave per 256-wide row. Safe for out == a.
// ---------------------------------------------------------------------------
__global__ __launch_bounds__(256) void ln_residual_kernel(
    const float* __restrict__ a, const float* __restrict__ bres,
    const float* __restrict__ g, const float* __restrict__ beta,
    float* __restrict__ out, int M) {
  const int wave = threadIdx.x >> 5, lane = threadIdx.x & 31;
  const int r = blockIdx.x * 8 + wave;
  if (r >= M) return;
  const float* pa = a + (size_t)r * DMODEL;
  const float* pb = bres + (size_t)r * DMODEL;
  float v[8]; float mu = 0.f;
#pragma unroll
  for (int c = 0; c < 8; ++c) { v[c] = pa[c * 32 + lane] + pb[c * 32 + lane]; mu += v[c]; }
  mu = warp_sum(mu) * (1.f / DMODEL);
  float var = 0.f;
#pragma unroll
  for (int c = 0; c < 8; ++c) { const float d = v[c] - mu; var += d * d; }
  var = warp_sum(var) * (1.f / DMODEL);
  const float rstd = rsqrtf(var + 1e-5f);
#pragma unroll
  for (int c = 0; c < 8; ++c) {
    const int col = c * 32 + lane;
    out[(size_t)r * DMODEL + col] = (v[c] - mu) * rstd * g[col] + beta[col];
  }
}

__global__ __launch_bounds__(256) void add_kernel(
    const float* __restrict__ a, const float* __restrict__ b,
    float* __restrict__ c, int n) {
  const int i = blockIdx.x * 256 + threadIdx.x;
  if (i < n) c[i] = a[i] + b[i];
}

__global__ __launch_bounds__(256) void concat_kernel(
    const float* __restrict__ x, const float* __restrict__ qe,
    float* __restrict__ c) {
  const int r = blockIdx.x;          // row
  const int t = threadIdx.x;         // 0..255
  c[(size_t)r * (2 * DMODEL) + t]           = x[(size_t)r * DMODEL + t];
  c[(size_t)r * (2 * DMODEL) + DMODEL + t]  = qe[(size_t)r * DMODEL + t];
}

// fc3 dot + sigmoid + masked BCE sum. One wave per row.
__global__ __launch_bounds__(256) void head_kernel(
    const float* __restrict__ h2, const float* __restrict__ w3,
    const float* __restrict__ b3, const float* __restrict__ target,
    float* __restrict__ out) {
  const int wave = threadIdx.x >> 5, lane = threadIdx.x & 31;
  const int r = blockIdx.x * 8 + wave;
  if (r >= ROWS) return;
  const float* p = h2 + (size_t)r * 256;
  float acc = 0.f;
#pragma unroll
  for (int c = 0; c < 8; ++c) acc += p[c * 32 + lane] * w3[c * 32 + lane];
  acc = warp_sum(acc);
  if (lane == 0) {
    const float pred = acc + b3[0];
    out[1 + r] = 1.f / (1.f + __expf(-pred));
    const float y = target[r];
    if (y > -0.9f) {
      const float bce = fmaxf(pred, 0.f) - pred * y + log1pf(__expf(-fabsf(pred)));
      atomicAdd(&out[0], bce);
      atomicAdd(&out[1 + ROWS], 1.0f);
    }
  }
}

__global__ void zero_scalars(float* out) { out[0] = 0.f; out[1 + ROWS] = 0.f; }

// ---------------------------------------------------------------------------
// Host side
// ---------------------------------------------------------------------------
struct LayerP {
  const float *q_w, *q_b, *k_w, *k_b, *v_w, *v_b, *o_w, *o_b, *gammas;
  const float *ln1_g, *ln1_b, *l1_w, *l1_b, *l2_w, *l2_b, *ln2_g, *ln2_b;
};

static LayerP load_layer(void* const* d_in, int base) {
  LayerP L;
  L.q_w = (const float*)d_in[base + 0];  L.q_b  = (const float*)d_in[base + 1];
  L.k_w = (const float*)d_in[base + 2];  L.k_b  = (const float*)d_in[base + 3];
  L.v_w = (const float*)d_in[base + 4];  L.v_b  = (const float*)d_in[base + 5];
  L.o_w = (const float*)d_in[base + 6];  L.o_b  = (const float*)d_in[base + 7];
  L.gammas = (const float*)d_in[base + 8];
  L.ln1_g = (const float*)d_in[base + 9];  L.ln1_b = (const float*)d_in[base + 10];
  L.l1_w  = (const float*)d_in[base + 11]; L.l1_b  = (const float*)d_in[base + 12];
  L.l2_w  = (const float*)d_in[base + 13]; L.l2_b  = (const float*)d_in[base + 14];
  L.ln2_g = (const float*)d_in[base + 15]; L.ln2_b = (const float*)d_in[base + 16];
  return L;
}

static void launch_gemm(const float* A, const float* W, const float* bias,
                        float* C, int M, int N, int K, int relu,
                        hipStream_t st) {
  dim3 grid(N / 64, M / 128);
  gemm_bf16_wmma<<<grid, 256, 0, st>>>(A, W, bias, C, M, N, K, relu);
}

struct WsBufs {
  float *bq, *bk, *bv, *ba, *bm, *bx, *bf;
};

static void run_tlayer(const LayerP& L, int mask_flag, int apply_pos,
                       const float* q, const float* k, const float* v,
                       float* out, const WsBufs& W, hipStream_t st) {
  launch_gemm(q, L.q_w, L.q_b, W.bq, ROWS, DMODEL, DMODEL, 0, st);
  launch_gemm(k, L.k_w, L.k_b, W.bk, ROWS, DMODEL, DMODEL, 0, st);
  launch_gemm(v, L.v_w, L.v_b, W.bv, ROWS, DMODEL, DMODEL, 0, st);
  attn_kernel<<<dim3(SEQ / 16, NHEADS, BS_), 256, 0, st>>>(
      W.bq, W.bk, W.bv, L.gammas, W.ba, mask_flag);
  launch_gemm(W.ba, L.o_w, L.o_b, W.bm, ROWS, DMODEL, DMODEL, 0, st);
  float* lnout = apply_pos ? W.bx : out;
  ln_residual_kernel<<<ROWS / 8, 256, 0, st>>>(q, W.bm, L.ln1_g, L.ln1_b,
                                               lnout, ROWS);
  if (apply_pos) {
    launch_gemm(W.bx, L.l1_w, L.l1_b, W.bf, ROWS, DFF, DMODEL, 1, st);
    launch_gemm(W.bf, L.l2_w, L.l2_b, W.bm, ROWS, DMODEL, DFF, 0, st);
    ln_residual_kernel<<<ROWS / 8, 256, 0, st>>>(W.bx, W.bm, L.ln2_g, L.ln2_b,
                                                 out, ROWS);
  }
}

extern "C" void kernel_launch(void* const* d_in, const int* in_sizes, int n_in,
                              void* d_out, int out_size, void* d_ws,
                              size_t ws_size, hipStream_t stream) {
  (void)in_sizes; (void)n_in; (void)out_size; (void)ws_size;
  // setup_inputs() dict order (recursive insertion order):
  const int*   q_data      = (const int*)d_in[0];
  const int*   qa_data     = (const int*)d_in[1];
  const float* target      = (const float*)d_in[2];
  const int*   graph       = (const int*)d_in[3];
  const float* q_embed     = (const float*)d_in[4];
  const float* qa_embed    = (const float*)d_in[5];
  const float* pos_embed   = (const float*)d_in[6];
  const float* graph_embed = (const float*)d_in[7];
  const LayerP b1_0 = load_layer(d_in, 8);
  const LayerP b2_0 = load_layer(d_in, 25);
  const LayerP b2_1 = load_layer(d_in, 42);
  const float* fc1_w = (const float*)d_in[59];
  const float* fc1_b = (const float*)d_in[60];
  const float* fc2_w = (const float*)d_in[61];
  const float* fc2_b = (const float*)d_in[62];
  const float* fc3_w = (const float*)d_in[63];
  const float* fc3_b = (const float*)d_in[64];

  float* out = (float*)d_out;
  float* ws  = (float*)d_ws;
  const size_t U = (size_t)ROWS * DMODEL;   // 1,048,576 floats
  float* q_e  = ws + 0 * U;
  float* x    = ws + 1 * U;
  float* y    = ws + 2 * U;
  float* g    = ws + 3 * U;
  float* xg   = ws + 4 * U;
  WsBufs W;
  W.bq = ws + 5 * U;  W.bk = ws + 6 * U;  W.bv = ws + 7 * U;
  W.ba = ws + 8 * U;  W.bm = ws + 9 * U;  W.bx = ws + 10 * U;
  W.bf = ws + 11 * U;                       // ROWS x DFF = 8U
  float* cbuf = ws + 19 * U;                // ROWS x 512 = 2U
  float* h1   = ws + 21 * U;                // ROWS x 512 = 2U
  float* h2   = ws + 23 * U;                // ROWS x 256 = 1U

  embed_kernel<<<ROWS, 256, 0, stream>>>(q_data, qa_data, graph, q_embed,
                                         qa_embed, pos_embed, graph_embed,
                                         q_e, x, y, g);

  // y = tlayer(b1_0, 1, y, y, y, apply_pos=True)
  run_tlayer(b1_0, 1, 1, y, y, y, y, W, stream);
  // x = tlayer(b2_0, 1, x, x, x, apply_pos=False)
  run_tlayer(b2_0, 1, 0, x, x, x, x, W, stream);
  // x = tlayer(b2_1, 0, x, x, y, apply_pos=True)
  run_tlayer(b2_1, 0, 1, x, x, y, x, W, stream);
  // x = tlayer(b2_0, 1, x+g, x+g, x+g, apply_pos=False)
  add_kernel<<<(ROWS * DMODEL) / 256, 256, 0, stream>>>(x, g, xg, ROWS * DMODEL);
  run_tlayer(b2_0, 1, 0, xg, xg, xg, x, W, stream);
  // x = tlayer(b2_1, 0, x+g, x+g, x, apply_pos=True)
  add_kernel<<<(ROWS * DMODEL) / 256, 256, 0, stream>>>(x, g, xg, ROWS * DMODEL);
  run_tlayer(b2_1, 0, 1, xg, xg, x, x, W, stream);

  // head: c = [x, q_e]; relu(fc1); relu(fc2); fc3 + sigmoid + masked BCE sum
  concat_kernel<<<ROWS, 256, 0, stream>>>(x, q_e, cbuf);
  launch_gemm(cbuf, fc1_w, fc1_b, h1, ROWS, FC1_, 2 * DMODEL, 1, stream);
  launch_gemm(h1, fc2_w, fc2_b, h2, ROWS, 256, FC1_, 1, stream);
  zero_scalars<<<1, 1, 0, stream>>>(out);
  head_kernel<<<ROWS / 8, 256, 0, stream>>>(h2, fc3_w, fc3_b, target, out);
}
